// Gumbel_Token_Selection_Block_Wrapper_40604620817112
// MI455X (gfx1250) — matrix operand order repaired and checked by reading, not compiled, not run on hardware
//
#include <hip/hip_runtime.h>
#include <math.h>

typedef __attribute__((ext_vector_type(16))) _Float16 v16h;
typedef __attribute__((ext_vector_type(8)))  float    v8f;

union FragH { v16h v; _Float16 e[16]; };
union FragF { v8f  v; float    e[8];  };
union H8    { uint4 u; _Float16 h[8]; };

// ---------------------------------------------------------------------------
// small helpers
// ---------------------------------------------------------------------------
__global__ __launch_bounds__(256) void cvt_f32_f16_kernel(const float* __restrict__ in,
                                                          _Float16* __restrict__ out, int n) {
    int i = blockIdx.x * 256 + threadIdx.x;
    if (i < n) out[i] = (_Float16)in[i];
}

__global__ __launch_bounds__(256) void zero_kernel(float* __restrict__ p, int n) {
    int i = blockIdx.x * 256 + threadIdx.x;
    if (i < n) p[i] = 0.0f;
}

// row-wise LayerNorm, one block per row. Safe for in==out (when OutT==float).
template <typename OutT>
__global__ __launch_bounds__(256) void ln_kernel(const float* __restrict__ in,
                                                 const float* __restrict__ w,
                                                 const float* __restrict__ b,
                                                 OutT* __restrict__ out, int D) {
    __shared__ float red[256];
    const int row = blockIdx.x, tid = threadIdx.x;
    const float* x = in + (size_t)row * D;
    float s = 0.0f;
    for (int i = tid; i < D; i += 256) s += x[i];
    red[tid] = s; __syncthreads();
    for (int st = 128; st > 0; st >>= 1) { if (tid < st) red[tid] += red[tid + st]; __syncthreads(); }
    const float mean = red[0] / (float)D;
    __syncthreads();
    float v = 0.0f;
    for (int i = tid; i < D; i += 256) { float d = x[i] - mean; v += d * d; }
    red[tid] = v; __syncthreads();
    for (int st = 128; st > 0; st >>= 1) { if (tid < st) red[tid] += red[tid + st]; __syncthreads(); }
    const float rstd = rsqrtf(red[0] / (float)D + 1e-5f);
    __syncthreads();
    OutT* y = out + (size_t)row * D;
    for (int i = tid; i < D; i += 256) y[i] = (OutT)((x[i] - mean) * rstd * w[i] + b[i]);
}

// ---------------------------------------------------------------------------
// WMMA GEMM: out[M,N] = epi( A[M,K](f16) @ W[N,K](f16)^T + bias[N] )
//   epi 0: plain   epi 1: exact GELU   epi 2: res + scale[col]*val
// outF (f32) and/or outH (f16) written if non-null.
// Macro tile 64(M) x 128(N); 8 waves; each wave owns one 16-wide N-strip
// -> 4 WMMA per K-step. Register-prefetched, LDS double-buffered staging
// (1 barrier / K-step). Sized to fit the 128-VGPR / 8-wave budget (no spills).
// M%64==0, N%128==0, K%32==0 guaranteed by caller.
// ---------------------------------------------------------------------------
__global__ __launch_bounds__(256)
void gemm_f16_wmma(const _Float16* __restrict__ A, int lda,
                   const _Float16* __restrict__ W, int K, int N,
                   const float* __restrict__ bias, int epi,
                   const float* __restrict__ res, const float* __restrict__ scale,
                   float* __restrict__ outF, _Float16* __restrict__ outH) {
    __shared__ _Float16 As[2][64][32];
    __shared__ _Float16 Ws[2][128][32];
    const int tid  = threadIdx.x;
    const int wid  = tid >> 5, lane = tid & 31;
    const int m0   = blockIdx.y * 64;
    const int n0   = blockIdx.x * 128;
    const int half = lane >> 4, l16 = lane & 15;

    FragF c[4];
#pragma unroll
    for (int j = 0; j < 4; ++j)
#pragma unroll
        for (int r = 0; r < 8; ++r) c[j].e[r] = 0.0f;

    // staging coordinates: 16B (8 halfs) per task; 32-bit offsets
    const int ar = tid >> 2, aseg = (tid & 3) * 8;      // A: 64x32 = 256 tasks
    const int arow = (m0 + ar) * lda + aseg;
    // W: 128x32 = 512 tasks, 2 per thread; only the row offsets stay live
    int wrow[2];
#pragma unroll
    for (int j = 0; j < 2; ++j) {
        const int t2 = tid + j * 256;
        wrow[j] = (n0 + (t2 >> 2)) * K + (t2 & 3) * 8;
    }

    // prefetch k-step 0 into registers
    uint4 ra = *(const uint4*)&A[arow];
    uint4 rw[2];
#pragma unroll
    for (int j = 0; j < 2; ++j) rw[j] = *(const uint4*)&W[wrow[j]];

    int buf = 0;
    for (int k0 = 0; k0 < K; k0 += 32, buf ^= 1) {
        // commit prefetched tile to LDS (recompute cheap store coords)
        *(uint4*)&As[buf][ar][aseg] = ra;
#pragma unroll
        for (int j = 0; j < 2; ++j) {
            const int t2 = tid + j * 256;
            *(uint4*)&Ws[buf][t2 >> 2][(t2 & 3) * 8] = rw[j];
        }

        // issue next tile's global loads (consumed next iteration)
        if (k0 + 32 < K) {
            ra = *(const uint4*)&A[arow + k0 + 32];
#pragma unroll
            for (int j = 0; j < 2; ++j) rw[j] = *(const uint4*)&W[wrow[j] + k0 + 32];
            __builtin_prefetch(&A[arow + k0 + 64], 0, 1);  // global_prefetch_b8
        }
        __syncthreads();

        FragH b;
        const int nc = wid * 16 + l16;
        *(uint4*)&b.e[0] = *(const uint4*)&Ws[buf][nc][half * 8];
        *(uint4*)&b.e[8] = *(const uint4*)&Ws[buf][nc][16 + half * 8];
#pragma unroll
        for (int j = 0; j < 4; ++j) {
            FragH a;
            const int mrow = j * 16 + l16;
            *(uint4*)&a.e[0] = *(const uint4*)&As[buf][mrow][half * 8];
            *(uint4*)&a.e[8] = *(const uint4*)&As[buf][mrow][16 + half * 8];
            c[j].v = __builtin_amdgcn_wmma_f32_16x16x32_f16(false, a.v, false, b.v,
                                                            (short)0, c[j].v, false, false);
        }
        // no trailing barrier: next iteration writes the other LDS buffer
    }

    const int col = n0 + wid * 16 + l16;
    const float bi = bias[col];
    const float sc = (epi == 2) ? scale[col] : 0.0f;
#pragma unroll
    for (int j = 0; j < 4; ++j) {
#pragma unroll
        for (int r = 0; r < 8; ++r) {
            const int row = m0 + j * 16 + half * 8 + r;
            float v = c[j].e[r] + bi;
            float o;
            if (epi == 1)      o = 0.5f * v * (1.0f + erff(v * 0.70710678118654752440f));
            else if (epi == 2) o = res[(size_t)row * N + col] + sc * v;
            else               o = v;
            if (outF) outF[(size_t)row * N + col] = o;
            if (outH) outH[(size_t)row * N + col] = (_Float16)o;
        }
    }
}

// ---------------------------------------------------------------------------
// WMMA attention: one block per (q-tile of 16, head, batch). qkv in f16,
// layout [B*197, 3*H*64] (q +0, k +768, v +1536; per-head offset h*64).
// Writes o16[B*197, 768] (f16); accumulates cls_att[B,197] (head-mean, q=0).
// ---------------------------------------------------------------------------
__global__ __launch_bounds__(256)
void attn_wmma(const _Float16* __restrict__ qkv, _Float16* __restrict__ o16,
               float* __restrict__ cls_att) {
    const int qt = blockIdx.x;   // 0..12
    const int h  = blockIdx.y;   // 0..11
    const int b  = blockIdx.z;   // 0..63
    const int q0 = qt * 16;
    const int tid = threadIdx.x, wid = tid >> 5, lane = tid & 31;
    const int half = lane >> 4, l16 = lane & 15;

    __shared__ float    S[16][224];
    __shared__ _Float16 P[16][224];
    __shared__ _Float16 Vt[64][224];   // V transposed: Vt[d][kv]
    __shared__ float    red[16][16];

    const size_t rs = 2304;
    const _Float16* Qb = qkv + (size_t)(b * 197) * rs + h * 64;
    const _Float16* Kb = Qb + 768;
    const _Float16* Vb = Qb + 1536;

    // stage V transposed into LDS: batch the 7 global loads, then scatter
    {
        H8 val[7];
#pragma unroll
        for (int t = 0; t < 7; ++t) {
            const int task = tid + t * 256;         // 224*8 = 1792 tasks
            const int kv = task >> 3, d0 = (task & 7) * 8;
            if (kv < 197) val[t].u = *(const uint4*)(Vb + (size_t)kv * rs + d0);
            else          { val[t].u.x = 0u; val[t].u.y = 0u; val[t].u.z = 0u; val[t].u.w = 0u; }
        }
#pragma unroll
        for (int t = 0; t < 7; ++t) {
            const int task = tid + t * 256;
            const int kv = task >> 3, d0 = (task & 7) * 8;
#pragma unroll
            for (int i = 0; i < 8; ++i) Vt[d0 + i][kv] = val[t].h[i];
        }
    }

    // Q fragments (A matrix, 16x64 split into two 16x32)
    FragH a0, a1;
    {
        int qr = q0 + l16; if (qr > 196) qr = 196;
        const _Float16* qp = Qb + (size_t)qr * rs;
        *(uint4*)&a0.e[0] = *(const uint4*)(qp + half * 8);
        *(uint4*)&a0.e[8] = *(const uint4*)(qp + 16 + half * 8);
        *(uint4*)&a1.e[0] = *(const uint4*)(qp + 32 + half * 8);
        *(uint4*)&a1.e[8] = *(const uint4*)(qp + 48 + half * 8);
    }

    // Phase 1: S = (Q K^T) * dh^-0.5 ; kv tiles round-robin over the 8 waves
    for (int t = wid; t < 14; t += 8) {
        FragH b0, b1;
        const int kv = t * 16 + l16;
        if (kv < 197) {
            const _Float16* kp = Kb + (size_t)kv * rs;
            *(uint4*)&b0.e[0] = *(const uint4*)(kp + half * 8);
            *(uint4*)&b0.e[8] = *(const uint4*)(kp + 16 + half * 8);
            *(uint4*)&b1.e[0] = *(const uint4*)(kp + 32 + half * 8);
            *(uint4*)&b1.e[8] = *(const uint4*)(kp + 48 + half * 8);
        } else {
#pragma unroll
            for (int i = 0; i < 16; ++i) { b0.e[i] = (_Float16)0.0f; b1.e[i] = (_Float16)0.0f; }
        }
        FragF c;
#pragma unroll
        for (int r = 0; r < 8; ++r) c.e[r] = 0.0f;
        c.v = __builtin_amdgcn_wmma_f32_16x16x32_f16(false, a0.v, false, b0.v, (short)0, c.v, false, false);
        c.v = __builtin_amdgcn_wmma_f32_16x16x32_f16(false, a1.v, false, b1.v, (short)0, c.v, false, false);
#pragma unroll
        for (int r = 0; r < 8; ++r) S[half * 8 + r][t * 16 + l16] = c.e[r] * 0.125f;
    }
    __syncthreads();

    // Phase 2: row softmax over 197 valid columns; probabilities into P (f16)
    {
        const int row = tid >> 4, cc = tid & 15;
        float m = -1e30f;
        for (int k = 0; k < 14; ++k) { int col = cc + 16 * k; if (col < 197) m = fmaxf(m, S[row][col]); }
        red[row][cc] = m; __syncthreads();
        if (cc == 0) { float mm = red[row][0]; for (int i = 1; i < 16; ++i) mm = fmaxf(mm, red[row][i]); red[row][0] = mm; }
        __syncthreads();
        const float rowmax = red[row][0];
        float sum = 0.0f;
        for (int k = 0; k < 14; ++k) {   // cache exp() back into S
            int col = cc + 16 * k;
            if (col < 197) { float e = expf(S[row][col] - rowmax); S[row][col] = e; sum += e; }
        }
        __syncthreads();
        red[row][cc] = sum; __syncthreads();
        if (cc == 0) { float ss = 0.0f; for (int i = 0; i < 16; ++i) ss += red[row][i]; red[row][0] = ss; }
        __syncthreads();
        const float inv = 1.0f / red[row][0];
        for (int k = 0; k < 14; ++k) {
            int col = cc + 16 * k;
            float p = (col < 197) ? S[row][col] * inv : 0.0f;
            P[row][col] = (_Float16)p;
            if (qt == 0 && row == 0 && col < 197)
                atomicAdd(&cls_att[b * 197 + col], p * (1.0f / 12.0f));
        }
    }
    __syncthreads();

    // Phase 3: O = P @ V ; waves 0..3 each own a 16-wide d-subtile
    if (wid < 4) {
        FragF c;
#pragma unroll
        for (int r = 0; r < 8; ++r) c.e[r] = 0.0f;
        const int dcol = wid * 16 + l16;
        for (int kk = 0; kk < 7; ++kk) {
            FragH a, bb;
            *(uint4*)&a.e[0]  = *(const uint4*)&P[l16][kk * 32 + half * 8];
            *(uint4*)&a.e[8]  = *(const uint4*)&P[l16][kk * 32 + 16 + half * 8];
            *(uint4*)&bb.e[0] = *(const uint4*)&Vt[dcol][kk * 32 + half * 8];
            *(uint4*)&bb.e[8] = *(const uint4*)&Vt[dcol][kk * 32 + 16 + half * 8];
            c.v = __builtin_amdgcn_wmma_f32_16x16x32_f16(false, a.v, false, bb.v, (short)0, c.v, false, false);
        }
#pragma unroll
        for (int r = 0; r < 8; ++r) {
            const int row = q0 + half * 8 + r;
            if (row < 197) o16[(size_t)(b * 197 + row) * 768 + h * 64 + dcol] = (_Float16)c.e[r];
        }
    }
}

// ---------------------------------------------------------------------------
// Gumbel scoring + top-49 selection (one block per batch element)
// ---------------------------------------------------------------------------
__device__ inline float uniform_hash(int b, int p) {
    unsigned x = (unsigned)(b * 196 + p) * 747796405u + 2891336453u;
    x = ((x >> ((x >> 28) + 4u)) ^ x) * 277803737u;
    x ^= x >> 22;
    float f = (float)(x >> 8) * (1.0f / 16777216.0f);
    return fminf(fmaxf(f, 1e-6f), 1.0f - 1e-6f);
}

__global__ __launch_bounds__(256)
void score_topk_kernel(const float* __restrict__ mcls, const float* __restrict__ ptri,
                       const float* __restrict__ cls_att,
                       const float* __restrict__ gp, const float* __restrict__ bp,
                       int* __restrict__ idx_out) {
    const int b = blockIdx.x, tid = threadIdx.x;
    __shared__ float sv[256];
    __shared__ float red[256];
    __shared__ int   redi[256];

    float s = 0.0f;
    if (tid < 196) {
        const float* mc = mcls + (size_t)b * 768;
        const float* pt = ptri + (size_t)(b * 197 + 1 + tid) * 768;
        float acc = 0.0f;
        for (int d = 0; d < 768; ++d) { float t = mc[d] * pt[d]; acc += t * t; }
        s = cls_att[b * 197 + 1 + tid] * (1.0f + sqrtf(acc));
    }
    red[tid] = (tid < 196) ? s : 0.0f; __syncthreads();
    for (int st = 128; st > 0; st >>= 1) { if (tid < st) red[tid] += red[tid + st]; __syncthreads(); }
    const float mu = red[0] / 196.0f; __syncthreads();
    red[tid] = (tid < 196) ? (s - mu) * (s - mu) : 0.0f; __syncthreads();
    for (int st = 128; st > 0; st >>= 1) { if (tid < st) red[tid] += red[tid + st]; __syncthreads(); }
    const float sd = sqrtf(red[0] / 195.0f); __syncthreads();

    float pert = -1e30f;
    if (tid < 196) {
        const float logits = gp[0] * (s - mu) / (sd + 1e-9f) + bp[0];
        const float u = uniform_hash(b, tid);
        pert = (logits - logf(-logf(u))) * 0.5f;   // / TAU
    }
    sv[tid] = pert; __syncthreads();

    for (int j = 0; j < 49; ++j) {
        red[tid] = sv[tid]; redi[tid] = tid; __syncthreads();
        for (int st = 128; st > 0; st >>= 1) {
            if (tid < st && red[tid + st] > red[tid]) { red[tid] = red[tid + st]; redi[tid] = redi[tid + st]; }
            __syncthreads();
        }
        if (tid == 0) { idx_out[b * 49 + j] = redi[0]; sv[redi[0]] = -1e30f; }
        __syncthreads();
    }
}

// out[b, 0, :] = x2[b, 0, :];  out[b, 1+j, :] = x2[b, 1+idx[b][j], :]
__global__ __launch_bounds__(256)
void gather_kernel(const float* __restrict__ x2, const int* __restrict__ idx,
                   float* __restrict__ out) {
    const int j = blockIdx.x, b = blockIdx.y, tid = threadIdx.x;
    const int src = (j == 0) ? 0 : 1 + idx[b * 49 + (j - 1)];
    const float* s = x2 + (size_t)(b * 197 + src) * 768;
    float* o = out + (size_t)(b * 50 + j) * 768;
    for (int e = tid; e < 768; e += 256) o[e] = s[e];
}

// ---------------------------------------------------------------------------
extern "C" void kernel_launch(void* const* d_in, const int* in_sizes, int n_in,
                              void* d_out, int out_size, void* d_ws, size_t ws_size,
                              hipStream_t stream) {
    (void)in_sizes; (void)n_in; (void)out_size; (void)ws_size;

    const float* x      = (const float*)d_in[0];
    const float* ln1_w  = (const float*)d_in[1];
    const float* ln1_b  = (const float*)d_in[2];
    const float* qkv_w  = (const float*)d_in[3];
    const float* qkv_b  = (const float*)d_in[4];
    const float* proj_w = (const float*)d_in[5];
    const float* proj_b = (const float*)d_in[6];
    const float* ls1    = (const float*)d_in[7];
    const float* ln2_w  = (const float*)d_in[8];
    const float* ln2_b  = (const float*)d_in[9];
    const float* fc1_w  = (const float*)d_in[10];
    const float* fc1_b  = (const float*)d_in[11];
    const float* fc2_w  = (const float*)d_in[12];
    const float* fc2_b  = (const float*)d_in[13];
    const float* ls2    = (const float*)d_in[14];
    const float* wu_w   = (const float*)d_in[15];
    const float* wu_b   = (const float*)d_in[16];
    const float* wtri_w = (const float*)d_in[17];
    const float* wtri_b = (const float*)d_in[18];
    const float* nu_w   = (const float*)d_in[19];
    const float* nu_b   = (const float*)d_in[20];
    const float* nt_w   = (const float*)d_in[21];
    const float* nt_b   = (const float*)d_in[22];
    const float* gamma  = (const float*)d_in[23];
    const float* beta   = (const float*)d_in[24];

    constexpr int B = 64, N = 197, D = 768, H = 12;
    constexpr int M = B * N;  // 12608 (multiple of 64)

    char* ws = (char*)d_ws;
    size_t off = 0;
    auto alloc = [&](size_t bytes) -> char* {
        char* p = ws + off;
        off += (bytes + 255) & ~(size_t)255;
        return p;
    };
    _Float16* w16_qkv  = (_Float16*)alloc((size_t)3 * D * D * 2);
    _Float16* w16_proj = (_Float16*)alloc((size_t)D * D * 2);
    _Float16* w16_fc1  = (_Float16*)alloc((size_t)4 * D * D * 2);
    _Float16* w16_fc2  = (_Float16*)alloc((size_t)4 * D * D * 2);
    _Float16* w16_wu   = (_Float16*)alloc((size_t)D * D * 2);
    _Float16* w16_wt   = (_Float16*)alloc((size_t)D * D * 2);
    _Float16* act16A = (_Float16*)alloc((size_t)M * D * 2);      // h16 / h2_16
    _Float16* act16B = (_Float16*)alloc((size_t)M * D * 2);      // o16 / x2_16
    _Float16* big16  = (_Float16*)alloc((size_t)M * 4 * D * 2);  // qkv16 / g16
    float* bufF1 = (float*)alloc((size_t)M * D * 4);             // x1 / x2 (in-place)
    float* bufF2 = (float*)alloc((size_t)M * D * 4);             // ptri
    float* mcls_pre = (float*)alloc((size_t)B * D * 4);
    float* mcls     = (float*)alloc((size_t)B * D * 4);
    float* clsA     = (float*)alloc((size_t)B * N * 4);
    int*   idxb     = (int*)alloc((size_t)B * 49 * 4);

    auto cvt = [&](const float* w, _Float16* o16, int n) {
        cvt_f32_f16_kernel<<<(n + 255) / 256, 256, 0, stream>>>(w, o16, n);
    };
    cvt(qkv_w,  w16_qkv,  3 * D * D);
    cvt(proj_w, w16_proj, D * D);
    cvt(fc1_w,  w16_fc1,  4 * D * D);
    cvt(fc2_w,  w16_fc2,  4 * D * D);
    cvt(wu_w,   w16_wu,   D * D);
    cvt(wtri_w, w16_wt,   D * D);
    zero_kernel<<<(B * N + 255) / 256, 256, 0, stream>>>(clsA, B * N);

    // h16 = LN1(x)
    ln_kernel<_Float16><<<M, 256, 0, stream>>>(x, ln1_w, ln1_b, act16A, D);
    // qkv16 = h @ qkv_w^T + qkv_b
    gemm_f16_wmma<<<dim3(3 * D / 128, M / 64), 256, 0, stream>>>(
        act16A, D, w16_qkv, D, 3 * D, qkv_b, 0, nullptr, nullptr, nullptr, big16);
    // o16 = attention(qkv16)  (+ cls_att)
    attn_wmma<<<dim3(13, H, B), 256, 0, stream>>>(big16, act16B, clsA);
    // x1 = x + ls1 * (o @ proj_w^T + proj_b)
    gemm_f16_wmma<<<dim3(D / 128, M / 64), 256, 0, stream>>>(
        act16B, D, w16_proj, D, D, proj_b, 2, x, ls1, bufF1, nullptr);
    // h2_16 = LN2(x1)
    ln_kernel<_Float16><<<M, 256, 0, stream>>>(bufF1, ln2_w, ln2_b, act16A, D);
    // g16 = gelu(h2 @ fc1_w^T + fc1_b)
    gemm_f16_wmma<<<dim3(4 * D / 128, M / 64), 256, 0, stream>>>(
        act16A, D, w16_fc1, D, 4 * D, fc1_b, 1, nullptr, nullptr, nullptr, big16);
    // x2 = x1 + ls2 * (g @ fc2_w^T + fc2_b)  (f32 in-place + f16 shadow)
    gemm_f16_wmma<<<dim3(D / 128, M / 64), 256, 0, stream>>>(
        big16, 4 * D, w16_fc2, 4 * D, D, fc2_b, 2, bufF1, ls2, bufF1, act16B);
    // m_cls = LN_nu(x2[:,0] @ wu_w^T + wu_b)   (A rows strided by 197*D)
    gemm_f16_wmma<<<dim3(D / 128, 1), 256, 0, stream>>>(
        act16B, N * D, w16_wu, D, D, wu_b, 0, nullptr, nullptr, mcls_pre, nullptr);
    ln_kernel<float><<<B, 256, 0, stream>>>(mcls_pre, nu_w, nu_b, mcls, D);
    // ptri = LN_nt(x2 @ wtri_w^T + wtri_b)
    gemm_f16_wmma<<<dim3(D / 128, M / 64), 256, 0, stream>>>(
        act16B, D, w16_wt, D, D, wtri_b, 0, nullptr, nullptr, bufF2, nullptr);
    ln_kernel<float><<<M, 256, 0, stream>>>(bufF2, nt_w, nt_b, bufF2, D);
    // scores -> gumbel -> top-49
    score_topk_kernel<<<B, 256, 0, stream>>>(mcls, bufF2, clsA, gamma, beta, idxb);
    // output: concat(cls, gathered patches)   (w_st == 1.0 in forward)
    gather_kernel<<<dim3(50, B), 256, 0, stream>>>(bufF1, idxb, (float*)d_out);
}
